// BiDetVGGBlock_34754875359419
// MI455X (gfx1250) — 8-bit matrix operand order repaired and checked by reading, compiled, not run on hardware
//
#include <hip/hip_runtime.h>
#include <hip/hip_bf16.h>

typedef int   v8i  __attribute__((ext_vector_type(8)));
typedef int   v16i __attribute__((ext_vector_type(16)));
typedef float v8f  __attribute__((ext_vector_type(8)));

#define NB   32
#define CCH  256
#define HH   56
#define WW   56
#define NPIX (NB * HH * WW)
#define PIXPERCH (NB * HH * WW)

#define FP8_P1 0x38
#define FP8_M1 0xB8
#define FP8_Z  0x00

#define XB_OFF   0ull
#define Y_OFF    25690112ull
#define WP_OFF   (Y_OFF + 51380224ull)
#define GSUM_OFF (WP_OFF + 589824ull)
#define GSQ_OFF  (GSUM_OFF + 2048ull)
#define PRM_OFF  (GSQ_OFF + 2048ull)

__device__ __forceinline__ unsigned char fp8_sign(float v) {
    return (v > 0.0f) ? (unsigned char)FP8_P1
         : (v < 0.0f) ? (unsigned char)FP8_M1
                      : (unsigned char)FP8_Z;
}

__global__ void k_zero_stats(unsigned long long* gsum, unsigned long long* gsq) {
    int t = threadIdx.x;
    gsum[t] = 0ull;
    gsq[t]  = 0ull;
}

__global__ void k_binarize_x(const float* __restrict__ x, unsigned char* __restrict__ xb) {
    __shared__ unsigned char sx[256 * 64];
    const int t  = threadIdx.x;
    const int p0 = blockIdx.x * 256;
    const int c0 = blockIdx.y * 64;

    const int pix = p0 + t;
    const int n   = pix / (HH * WW);
    const int rem = pix - n * (HH * WW);
    const float* xp = x + (size_t)n * CCH * HH * WW + (size_t)c0 * HH * WW + rem;
#pragma unroll 4
    for (int j = 0; j < 64; ++j) {
        float v = xp[(size_t)j * (HH * WW)];
        sx[t * 64 + j] = fp8_sign(v);
    }
    __syncthreads();

    const uint4* ls = (const uint4*)sx;
#pragma unroll
    for (int k = 0; k < 4; ++k) {
        int gi  = t + k * 256;
        int off = gi * 16;
        int pl  = off >> 6;
        uint4* dst = (uint4*)(xb + (size_t)(p0 + pl) * CCH + c0 + (off & 63));
        *dst = ls[gi];
    }
}

__global__ void k_pack_w(const float* __restrict__ w, unsigned char* __restrict__ wp) {
    const int gid  = blockIdx.x * 256 + threadIdx.x;
    const int b    = gid & 63;
    const int lane = (gid >> 6) & 31;
    const int nt   = (gid >> 11) & 15;
    const int j    = gid >> 15;
    const int half = b >> 5;
    const int jj   = b & 31;
    const int ks   = j * 2 + half;
    const int tap  = ks >> 2;
    const int cc   = ks & 3;
    const int kh   = tap / 3;
    const int kw   = tap - kh * 3;
    const int hi   = lane >> 4;
    const int nl   = lane & 15;
    const int ch   = nt * 16 + nl;
    const int kloc = (jj & 7) + ((jj >> 3) << 4) + (hi << 3);
    const int cin  = cc * 64 + kloc;
    float v = w[(((size_t)ch * CCH + cin) * 3 + kh) * 3 + kw];
    wp[gid] = fp8_sign(v);
}

__global__ void __launch_bounds__(256)
k_conv_wmma(const unsigned char* __restrict__ xb, const unsigned char* __restrict__ wp,
            short* __restrict__ y, unsigned long long* gsum, unsigned long long* gsq) {
    __shared__ int ssum[CCH];
    __shared__ int ssq[CCH];
    __shared__ unsigned char xs[3 * 64 * CCH];

    const int tid = threadIdx.x;
    const int n   = blockIdx.x / HH;
    const int h   = blockIdx.x % HH;

    {
        uint4 z = make_uint4(0u, 0u, 0u, 0u);
        uint4* xz = (uint4*)xs;
#pragma unroll
        for (int i = 0; i < 12; ++i) xz[tid + i * 256] = z;
        ssum[tid] = 0;
        ssq[tid]  = 0;
    }
    for (int r = 0; r < 3; ++r) {
        int hh = h - 1 + r;
        if (hh >= 0 && hh < HH) {
            const uint4* src = (const uint4*)(xb + (size_t)((n * HH + hh) * WW) * CCH);
            uint4* dst = (uint4*)(xs + r * 64 * CCH + CCH);
            for (int i = tid; i < (WW * CCH) / 16; i += 256) dst[i] = src[i];
        }
    }
    __syncthreads();

    const int lane   = tid & 31;
    const int wave   = tid >> 5;
    const int wave_m = wave & 3;
    const int wave_n = wave >> 2;
    const int w0     = wave_m * 16;
    const int nl     = lane & 15;
    const int hi     = lane >> 4;

    v8f acc[8];
#pragma unroll
    for (int i = 0; i < 8; ++i) acc[i] = (v8f){0.f, 0.f, 0.f, 0.f, 0.f, 0.f, 0.f, 0.f};

    for (int tap = 0; tap < 9; ++tap) {
        const int kh = tap / 3;
        const int kw = tap - kh * 3;
        const int abase = (kh * 64 + (w0 + nl + kw)) * CCH + hi * 8;
#pragma unroll
        for (int half = 0; half < 2; ++half) {
            const unsigned long long* ap =
                (const unsigned long long*)(xs + abase + half * 128);
            union { unsigned long long q[8]; v16i v; } ua;
            ua.q[0] = ap[0];
            ua.q[1] = ap[2];
            ua.q[2] = ap[4];
            ua.q[3] = ap[6];
            ua.q[4] = ap[8];
            ua.q[5] = ap[10];
            ua.q[6] = ap[12];
            ua.q[7] = ap[14];
            const int j = tap * 2 + half;
#pragma unroll
            for (int nt = 0; nt < 8; ++nt) {
                const int ct = wave_n * 8 + nt;
                const uint4* bp = (const uint4*)(wp +
                    ((((size_t)j * 16 + ct) * 32 + lane) << 6));
                union { uint4 u[4]; v16i v; } ub;
                ub.u[0] = bp[0];
                ub.u[1] = bp[1];
                ub.u[2] = bp[2];
                ub.u[3] = bp[3];
                acc[nt] = __builtin_amdgcn_wmma_f32_16x16x128_fp8_fp8(
                    ua.v, ub.v, (short)0, acc[nt], false, false);
            }
        }
    }

    short* yrow = y + (size_t)((n * HH + h) * WW) * CCH;
    const int mb = hi * 8;
#pragma unroll
    for (int nt = 0; nt < 8; ++nt) {
        const int ch = wave_n * 128 + nt * 16 + nl;
        int s = 0, q = 0;
#pragma unroll
        for (int v = 0; v < 8; ++v) {
            const int m  = mb + v;
            const int w  = w0 + m;
            const int vv = (int)acc[nt][v];
            if (w < WW) {
                yrow[(size_t)w * CCH + ch] = (short)vv;
                s += vv;
                q += vv * vv;
            }
        }
        atomicAdd(&ssum[ch], s);
        atomicAdd(&ssq[ch], q);
    }
    __syncthreads();

    atomicAdd(&gsum[tid], (unsigned long long)(long long)ssum[tid]);
    atomicAdd(&gsq[tid],  (unsigned long long)(long long)ssq[tid]);
}

__global__ void k_finalize(const unsigned long long* gsum, const unsigned long long* gsq,
                           const float* __restrict__ gamma, const float* __restrict__ beta,
                           float* __restrict__ prm) {
    int c = threadIdx.x;
    long long s = (long long)gsum[c];
    double mean = (double)s / (double)PIXPERCH;
    double var  = (double)gsq[c] / (double)PIXPERCH - mean * mean;
    double inv  = 1.0 / sqrt(var + 1e-5);
    float sc = gamma[c] * (float)inv;
    prm[c]       = sc;
    prm[256 + c] = beta[c] - (float)mean * sc;
}

__global__ void __launch_bounds__(256)
k_bn_residual(const float* __restrict__ x, const short* __restrict__ y,
              const float* __restrict__ prm, float* __restrict__ out) {
    __shared__ short ys[WW * CCH];
    __shared__ float ssc[CCH];
    __shared__ float ssh[CCH];
    const int tid = threadIdx.x;
    ssc[tid] = prm[tid];
    ssh[tid] = prm[256 + tid];

    const uint4* src = (const uint4*)(y + (size_t)blockIdx.x * (WW * CCH));
    uint4* dst = (uint4*)ys;
    for (int i = tid; i < (WW * CCH * 2) / 16; i += 256) dst[i] = src[i];
    __syncthreads();

    const int n = blockIdx.x / HH;
    const int h = blockIdx.x % HH;
    const size_t obase = (size_t)n * CCH * HH * WW + (size_t)h * WW;
    for (int idx = tid; idx < WW * CCH; idx += 256) {
        const int c = idx / WW;
        const int w = idx - c * WW;
        const float yv = (float)ys[w * CCH + c];
        const size_t o = obase + (size_t)c * (HH * WW) + w;
        out[o] = yv * ssc[c] + ssh[c] + x[o];
    }
}

extern "C" void kernel_launch(void* const* d_in, const int* in_sizes, int n_in,
                              void* d_out, int out_size, void* d_ws, size_t ws_size,
                              hipStream_t stream) {
    const float* x     = (const float*)d_in[0];
    const float* wgt   = (const float*)d_in[1];
    const float* gamma = (const float*)d_in[2];
    const float* beta  = (const float*)d_in[3];
    float* out = (float*)d_out;

    unsigned char* ws = (unsigned char*)d_ws;
    unsigned char* xb = ws + XB_OFF;
    short*         y  = (short*)(ws + Y_OFF);
    unsigned char* wp = ws + WP_OFF;
    unsigned long long* gsum = (unsigned long long*)(ws + GSUM_OFF);
    unsigned long long* gsq  = (unsigned long long*)(ws + GSQ_OFF);
    float*              prm  = (float*)(ws + PRM_OFF);

    k_zero_stats<<<1, 256, 0, stream>>>(gsum, gsq);
    k_binarize_x<<<dim3(NPIX / 256, 4), 256, 0, stream>>>(x, xb);
    k_pack_w<<<2304, 256, 0, stream>>>(wgt, wp);
    k_conv_wmma<<<NB * HH, 256, 0, stream>>>(xb, wp, y, gsum, gsq);
    k_finalize<<<1, 256, 0, stream>>>(gsum, gsq, gamma, beta, prm);
    k_bn_residual<<<NB * HH, 256, 0, stream>>>(x, y, prm, out);
}
